// AxialDiscriminator_69647189672116
// MI455X (gfx1250) — compile-verified, matrix operand order checked by reading, not run on hardware
//
#include <hip/hip_runtime.h>
#include <hip/hip_bf16.h>
#include <stdint.h>

// Problem constants (from the reference)
#define D_  8
#define H_  4
#define E_  4
#define NL_ 4
#define B_  4
#define S_  20
#define IM_ 64

typedef __attribute__((ext_vector_type(16))) _Float16 v16h;
typedef __attribute__((ext_vector_type(8)))  float    v8f;
typedef unsigned int u32x4 __attribute__((ext_vector_type(4)));
typedef int          i32x4 __attribute__((ext_vector_type(4)));
typedef int          i32x8 __attribute__((ext_vector_type(8)));

#if __has_builtin(__builtin_amdgcn_tensor_load_to_lds) && \
    __has_builtin(__builtin_amdgcn_s_wait_tensorcnt)
#define HAVE_TDM 1
#else
#define HAVE_TDM 0
#endif

// ---------------------------------------------------------------------------
// Kernel 1: embedding  z0[b,s,y,x,d] = relu(x*conv_w[d]) + pos_s + pos_h + pos_w
// ---------------------------------------------------------------------------
__global__ __launch_bounds__(256) void embed_kernel(
    const float* __restrict__ x, const float* __restrict__ conv_w,
    const float* __restrict__ pos_s, const float* __restrict__ pos_h,
    const float* __restrict__ pos_w, float* __restrict__ z0) {
  int idx = blockIdx.x * blockDim.x + threadIdx.x;
  const int N = B_ * S_ * IM_ * IM_ * D_;
  if (idx >= N) return;
  int d  = idx % D_;
  int p  = idx / D_;          // ((b*S+s)*IM+y)*IM+xx : matches x's layout
  int xx = p % IM_;
  int py = p / IM_;
  int y  = py % IM_;
  int s  = (py / IM_) % S_;
  float val = x[p] * conv_w[d];
  val = fmaxf(val, 0.f);
  val += pos_s[s * D_ + d] + pos_h[d * IM_ + y] + pos_w[d * IM_ + xx];
  z0[idx] = val;
}

// ---------------------------------------------------------------------------
// TDM: one tensor_load_to_lds moves an L x 8-float tile (row stride in floats)
// from global into LDS. Descriptor per CDNA5 ISA.
// All inputs are wave-uniform. Tracked by TENSORcnt.
// ---------------------------------------------------------------------------
__device__ __forceinline__ void tdm_load_rows(unsigned lds_off, const float* gptr,
                                              int nrows, int stride_f) {
#if HAVE_TDM
  unsigned long long ga = (unsigned long long)(uintptr_t)gptr;
  u32x4 g0;
  g0[0] = 1u;                                            // count=1 (valid user D#)
  g0[1] = lds_off;                                       // lds_addr (bytes)
  g0[2] = (unsigned)(ga & 0xFFFFFFFFu);                  // global_addr[31:0]
  g0[3] = (unsigned)((ga >> 32) & 0x01FFFFFFu)           // global_addr[56:32]
          | (2u << 30);                                  // type=2 ("image")
  i32x8 g1;
  g1[0] = (2 << 16);                                     // data_size=4B; no pad/iter
  g1[1] = (int)((unsigned)(D_ & 0xFFFF) << 16);          // tensor_dim0[15:0]
  g1[2] = (int)(((unsigned)nrows & 0xFFFFu) << 16);      // dim0[31:16]=0 | tensor_dim1[15:0]
  g1[3] = (int)((unsigned)D_ << 16);                     // dim1[31:16]=0 | tile_dim0=8
  g1[4] = (int)((unsigned)nrows & 0xFFFFu);              // tile_dim1=L | tile_dim2=0
  g1[5] = stride_f;                                      // tensor_dim0_stride[31:0]
  g1[6] = 0;                                             // stride[47:32] | dim1_stride lo
  g1[7] = 0;
  i32x4 g2 = {0, 0, 0, 0};                               // 2D tensor: dims 2/3 unused
  i32x4 g3 = {0, 0, 0, 0};
#if __clang_major__ >= 23
  i32x8 g4 = {0, 0, 0, 0, 0, 0, 0, 0};
  __builtin_amdgcn_tensor_load_to_lds(g0, g1, g2, g3, g4, 0);
#else
  __builtin_amdgcn_tensor_load_to_lds(g0, g1, g2, g3, 0);
#endif
#endif
}

// ---------------------------------------------------------------------------
// Kernel 2: one wave per axial row. TDM LDS staging + WMMA projections +
// VALU online softmax. f16 A/B fragments per CDNA5 16-bit A layout:
//   lane n   (<16): M=n, elems 0..7 = K=0..7, elems 8..15 = K=16..23
//   lane n+16     : M=n, elems 0..7 = K=8..15, elems 8..15 = K=24..31
// B (K x N) mirrored: lane n holds column N=n, same K packing.
// C/D f32: VGPR r -> M = r + (lane>=16 ? 8 : 0), N = lane & 15.
// ---------------------------------------------------------------------------
__global__ __launch_bounds__(32) void axial_attn_kernel(
    const float* __restrict__ zin, float* __restrict__ zout,
    const float* __restrict__ Wq, const float* __restrict__ Wkv,
    const float* __restrict__ Wo, const float* __restrict__ bo,
    int axis, int accumulate) {
  __shared__ float sZ[64 * D_];
  __shared__ float sQ[64 * 16];
  __shared__ float sK[64 * 16];
  __shared__ float sV[64 * 16];
  __shared__ float sO[64 * 16];

  const int lane = threadIdx.x;
  const int r = blockIdx.x;
  int L, stride, base;
  if (axis == 0) {                       // attend over S
    L = S_; stride = IM_ * IM_ * D_;
    int b = r / (IM_ * IM_); int rem = r % (IM_ * IM_);
    int y = rem / IM_, xx = rem % IM_;
    base = (((b * S_) * IM_ + y) * IM_ + xx) * D_;
  } else if (axis == 1) {                // attend over y
    L = IM_; stride = IM_ * D_;
    int b = r / (S_ * IM_); int s = (r / IM_) % S_; int xx = r % IM_;
    base = (((b * S_ + s) * IM_) * IM_ + xx) * D_;
  } else {                               // attend over x
    L = IM_; stride = D_;
    int b = r / (S_ * IM_); int s = (r / IM_) % S_; int y = r % IM_;
    base = (((b * S_ + s) * IM_ + y) * IM_) * D_;
  }
  const int ntiles = (L + 15) >> 4;
  const int padded = ntiles * 16;

#if HAVE_TDM
  // DMA the whole L x 8 row tile into LDS in one TDM op; the engine handles
  // the strided gather (axis 0: 20 x 32B segments 128KB apart).
  tdm_load_rows((unsigned)(uintptr_t)&sZ[0], zin + base, L, stride);
  // Zero-pad rows [L, padded) (disjoint from the TDM tile) while DMA runs.
  for (int i = lane; i < padded * D_; i += 32)
    if (i / D_ >= L) sZ[i] = 0.f;
  for (int i = lane; i < padded * 16; i += 32)
    if ((i >> 4) >= L) sO[i] = 0.f;
  __builtin_amdgcn_s_wait_tensorcnt(0);
#else
  for (int i = lane; i < padded * D_; i += 32) {
    int row = i / D_;
    sZ[i] = (row < L) ? zin[base + row * stride + (i % D_)] : 0.f;
  }
  for (int i = lane; i < padded * 16; i += 32)
    if ((i >> 4) >= L) sO[i] = 0.f;
#endif
  __syncthreads();

  const int  lo = lane & 15;
  const bool hi = lane >= 16;

  // ---- q/k/v projections: (16 x 8) @ (8 x 16|16) as 16x16x32 f16 WMMA ----
  for (int t = 0; t < ntiles; ++t) {
    v16h a = {};
#pragma unroll
    for (int kk = 0; kk < 8; ++kk) {
      float zv = sZ[(t * 16 + lo) * D_ + kk];      // K = kk (0..7); K>=8 is pad
      a[kk] = hi ? (_Float16)0.f : (_Float16)zv;
    }
    v16h bq = {}, bk = {}, bv = {};
#pragma unroll
    for (int kk = 0; kk < 8; ++kk) {
      float wq = Wq[kk * 16 + lo];
      float wk = Wkv[kk * 32 + lo];
      float wv = Wkv[kk * 32 + 16 + lo];
      bq[kk] = hi ? (_Float16)0.f : (_Float16)wq;
      bk[kk] = hi ? (_Float16)0.f : (_Float16)wk;
      bv[kk] = hi ? (_Float16)0.f : (_Float16)wv;
    }
    v8f cz = {};
    v8f cq = __builtin_amdgcn_wmma_f32_16x16x32_f16(false, a, false, bq, (short)0, cz, false, false);
    v8f ck = __builtin_amdgcn_wmma_f32_16x16x32_f16(false, a, false, bk, (short)0, cz, false, false);
    v8f cv = __builtin_amdgcn_wmma_f32_16x16x32_f16(false, a, false, bv, (short)0, cz, false, false);
#pragma unroll
    for (int rr = 0; rr < 8; ++rr) {
      int row = t * 16 + rr + (hi ? 8 : 0);
      sQ[row * 16 + lo] = cq[rr];
      sK[row * 16 + lo] = ck[rr];
      sV[row * 16 + lo] = cv[rr];
    }
  }
  __syncthreads();

  // ---- per-row attention, online softmax (K=4 per head -> VALU) ----
  for (int i = lane; i < L; i += 32) {
    float qi[16];
#pragma unroll
    for (int c = 0; c < 16; ++c) qi[c] = sQ[i * 16 + c];
    float m[H_], ssum[H_], acc[16];
#pragma unroll
    for (int h = 0; h < H_; ++h) { m[h] = -3.4e38f; ssum[h] = 0.f; }
#pragma unroll
    for (int c = 0; c < 16; ++c) acc[c] = 0.f;
    for (int j = 0; j < L; ++j) {
#pragma unroll
      for (int h = 0; h < H_; ++h) {
        float dot = 0.f;
#pragma unroll
        for (int e = 0; e < E_; ++e)
          dot += qi[h * E_ + e] * sK[j * 16 + h * E_ + e];
        dot *= 0.5f;                              // E^-0.5
        float mn   = fmaxf(m[h], dot);
        float corr = __expf(m[h] - mn);
        float p    = __expf(dot - mn);
        ssum[h] = ssum[h] * corr + p;
#pragma unroll
        for (int e = 0; e < E_; ++e)
          acc[h * E_ + e] = acc[h * E_ + e] * corr + p * sV[j * 16 + h * E_ + e];
        m[h] = mn;
      }
    }
#pragma unroll
    for (int h = 0; h < H_; ++h) {
      float inv = 1.f / ssum[h];
#pragma unroll
      for (int e = 0; e < E_; ++e)
        sO[i * 16 + h * E_ + e] = acc[h * E_ + e] * inv;
    }
  }
  __syncthreads();

  // ---- output projection: (16 x 16) @ (16 x 8) as 16x16x32 f16 WMMA ----
  for (int t = 0; t < ntiles; ++t) {
    v16h a = {};
#pragma unroll
    for (int kk = 0; kk < 8; ++kk)                 // lane<16: K=0..7; lane>=16: K=8..15
      a[kk] = (_Float16)sO[(t * 16 + lo) * 16 + (hi ? 8 : 0) + kk];
    v16h bw = {};
#pragma unroll
    for (int kk = 0; kk < 8; ++kk) {
      float w = (lo < D_) ? Wo[((hi ? 8 : 0) + kk) * D_ + lo] : 0.f;
      bw[kk] = (_Float16)w;
    }
    v8f cz = {};
    v8f co = __builtin_amdgcn_wmma_f32_16x16x32_f16(false, a, false, bw, (short)0, cz, false, false);
#pragma unroll
    for (int rr = 0; rr < 8; ++rr) {
      int row = t * 16 + rr + (hi ? 8 : 0);
      if (row < L && lo < D_) {
        float val = co[rr] + bo[lo];
        int idx = base + row * stride + lo;
        if (accumulate) zout[idx] += val; else zout[idx] = val;
      }
    }
  }
}

// ---------------------------------------------------------------------------
// Kernel 3: residual + max over S + dot(Wc) + sigmoid, one block per batch.
// ---------------------------------------------------------------------------
__global__ __launch_bounds__(256) void final_kernel(
    const float* __restrict__ z, const float* __restrict__ z0,
    const float* __restrict__ Wc, const float* __restrict__ bc,
    float* __restrict__ out) {
  __shared__ float red[256];
  const int b = blockIdx.x;
  const int tid = threadIdx.x;
  float partial = 0.f;
  for (int p = tid; p < D_ * IM_ * IM_; p += 256) {   // p = d*IM*IM + y*IM + x (flat order)
    int d = p / (IM_ * IM_);
    int rem = p % (IM_ * IM_);
    int y = rem / IM_, xx = rem % IM_;
    int idx0 = (((b * S_) * IM_ + y) * IM_ + xx) * D_ + d;
    float mx = -3.4e38f;
    for (int s = 0; s < S_; ++s) {
      int idx = idx0 + s * (IM_ * IM_ * D_);
      float v = z[idx] + z0[idx];
      mx = fmaxf(mx, v);
    }
    partial += mx * Wc[p];
  }
  red[tid] = partial;
  __syncthreads();
  for (int off = 128; off > 0; off >>= 1) {
    if (tid < off) red[tid] += red[tid + off];
    __syncthreads();
  }
  if (tid == 0) {
    float t = red[0] + bc[0];
    out[b] = 1.f / (1.f + __expf(-t));
  }
}

// ---------------------------------------------------------------------------
extern "C" void kernel_launch(void* const* d_in, const int* in_sizes, int n_in,
                              void* d_out, int out_size, void* d_ws, size_t ws_size,
                              hipStream_t stream) {
  const float* x      = (const float*)d_in[0];
  const float* conv_w = (const float*)d_in[1];
  const float* pos_s  = (const float*)d_in[2];
  const float* pos_h  = (const float*)d_in[3];
  const float* pos_w  = (const float*)d_in[4];
  const float* Wq     = (const float*)d_in[5];
  const float* Wkv    = (const float*)d_in[6];
  const float* Wo     = (const float*)d_in[7];
  const float* bo     = (const float*)d_in[8];
  const float* Wc     = (const float*)d_in[9];
  const float* bc     = (const float*)d_in[10];
  float* out = (float*)d_out;

  const int N = B_ * S_ * IM_ * IM_ * D_;       // 2,621,440 floats
  float* z0 = (float*)d_ws;                     // 3 buffers x ~10.5 MB
  float* zA = z0 + N;
  float* zB = zA + N;

  embed_kernel<<<(N + 255) / 256, 256, 0, stream>>>(x, conv_w, pos_s, pos_h, pos_w, z0);

  const float* cur = z0;
  float* bufs[2] = {zA, zB};
  for (int l = 0; l < NL_; ++l) {
    float* nxt = bufs[l & 1];
    for (int a = 0; a < 3; ++a) {
      int nrows = (a == 0) ? (B_ * IM_ * IM_) : (B_ * S_ * IM_);
      int w = l * 3 + a;
      axial_attn_kernel<<<nrows, 32, 0, stream>>>(
          cur, nxt,
          Wq + w * D_ * 16, Wkv + w * D_ * 32, Wo + w * 16 * D_, bo + w * D_,
          a, (a > 0) ? 1 : 0);
    }
    cur = nxt;
  }
  final_kernel<<<B_, 256, 0, stream>>>(cur, z0, Wc, bc, out);
}
